// Int8IMEGPT2MLP_14637248544857
// MI455X (gfx1250) — 8-bit matrix operand order repaired and checked by reading, compiled, not run on hardware
//
#include <hip/hip_runtime.h>
#include <hip/hip_bf16.h>

typedef __attribute__((ext_vector_type(8))) int v8i;

#define Dm 768
#define Fm 3072
#define ROWS 64
#define NBLK ((4 * 4096) / ROWS)

#define S_FC_IN   0.02f
#define S_G_IN    0.05f
#define ZP_G_IN   (-10)
#define S_G_OUT   0.01f
#define W1_S      0.01f
#define W2_S      0.01f
#define ZP_G_OUT  (-120)
#define M1f       (S_FC_IN * W1_S / S_G_IN)
#define OUT_SCALE (S_G_OUT * W2_S)

#define LUT_BYTES 1024
#define QS_BYTES  (ROWS * Dm)
#define GS_BYTES  (ROWS * Fm)
#define SMEM_BYTES (LUT_BYTES + QS_BYTES + GS_BYTES)

__device__ __forceinline__ int clamp8(int v) {
  return v < -128 ? -128 : (v > 127 ? 127 : v);
}


__global__ __launch_bounds__(256) void pack_w1t(const int* __restrict__ W1,
                                                signed char* __restrict__ w1t) {
  int idx = blockIdx.x * blockDim.x + threadIdx.x;
  int f = idx / Dm, k = idx % Dm;
  w1t[f * Dm + k] = (signed char)W1[idx];
}

__global__ __launch_bounds__(256) void pack_w2t(const int* __restrict__ W2,
                                                signed char* __restrict__ w2t) {
  int idx = blockIdx.x * blockDim.x + threadIdx.x;
  int d = idx / Fm, f = idx % Fm;
  w2t[d * Fm + f] = (signed char)W2[idx];
}

__global__ __launch_bounds__(256) void colsum_w2(const int* __restrict__ W2,
                                                 int* __restrict__ colsum) {
  __shared__ int red[256];
  const int d = blockIdx.x;
  int s = 0;
  for (int i = threadIdx.x; i < Fm; i += 256) s += W2[d * Fm + i];
  red[threadIdx.x] = s;
  __syncthreads();
  for (int off = 128; off > 0; off >>= 1) {
    if (threadIdx.x < off) red[threadIdx.x] += red[threadIdx.x + off];
    __syncthreads();
  }
  if (threadIdx.x == 0) colsum[d] = red[0];
}

__global__ __launch_bounds__(256) void mlp_fused(
    const float* __restrict__ hidden, const int* __restrict__ b1,
    const float* __restrict__ b2, const int* __restrict__ lut,
    const signed char* __restrict__ w1t, const signed char* __restrict__ w2t,
    const int* __restrict__ colsum2, float* __restrict__ out) {
  extern __shared__ char smem[];
  int* lutS = (int*)smem;
  signed char* qS = (signed char*)(smem + LUT_BYTES);
  signed char* gS = (signed char*)(smem + LUT_BYTES + QS_BYTES);

  const int tid = threadIdx.x;
  const int wave = tid >> 5;
  const int lane = tid & 31;
  const int m = lane & 15;
  const int hi = lane >> 4;
  const int rowBase = blockIdx.x * ROWS;

  lutS[tid] = lut[tid];

  {
    const float4* hb = (const float4*)(hidden + rowBase * Dm);
    int* qS4 = (int*)qS;
#pragma unroll 4
    for (int it = 0; it < (ROWS * Dm / 4) / 256; ++it) {
      int v = it * 256 + tid;
      float4 x = hb[v];
      int q0 = clamp8((int)rintf(x.x * (1.0f / S_FC_IN)));
      int q1 = clamp8((int)rintf(x.y * (1.0f / S_FC_IN)));
      int q2 = clamp8((int)rintf(x.z * (1.0f / S_FC_IN)));
      int q3 = clamp8((int)rintf(x.w * (1.0f / S_FC_IN)));
      qS4[v] = (q0 & 255) | ((q1 & 255) << 8) | ((q2 & 255) << 16) | ((q3 & 255) << 24);
    }
  }
  __syncthreads();

  for (int grp = 0; grp < 8; ++grp) {
    const int tileBase = wave * 24 + grp * 3;
    v8i acc[3][4] = {};
    for (int kc = 0; kc < 12; ++kc) {
      const int kb = kc * 64;
      v8i a[4];
#pragma unroll
      for (int mt = 0; mt < 4; ++mt) {
#pragma unroll
        for (int c = 0; c < 4; ++c) {
          int2 t = *(const int2*)(qS + (mt * 16 + m) * Dm + kb + c * 16 + hi * 8);
          a[mt][2 * c] = t.x;
          a[mt][2 * c + 1] = t.y;
        }
      }
#pragma unroll
      for (int i = 0; i < 3; ++i) {
        const int n0 = (tileBase + i) * 16;
        v8i bm;
#pragma unroll
        for (int c = 0; c < 4; ++c) {
          int2 t = *(const int2*)(w1t + (n0 + m) * Dm + kb + c * 16 + hi * 8);
          bm[2 * c] = t.x;
          bm[2 * c + 1] = t.y;
        }
#pragma unroll
        for (int mt = 0; mt < 4; ++mt)
          acc[i][mt] = __builtin_amdgcn_wmma_i32_16x16x64_iu8(
              true, a[mt], true, bm, acc[i][mt], false, false);
      }
    }
#pragma unroll
    for (int i = 0; i < 3; ++i) {
      const int col = (tileBase + i) * 16 + m;
      const int bias = b1[col];
#pragma unroll
      for (int mt = 0; mt < 4; ++mt) {
#pragma unroll
        for (int j = 0; j < 8; ++j) {
          int t = (int)rintf((float)(acc[i][mt][j] + bias) * M1f) + ZP_G_IN;
          t = clamp8(t);
          gS[(mt * 16 + j + 8 * hi) * Fm + col] = (signed char)lutS[t + 128];
        }
      }
    }
  }
  __syncthreads();

  for (int grp = 0; grp < 2; ++grp) {
    const int tileBase = wave * 6 + grp * 3;
    v8i acc[3][4] = {};
    for (int kc = 0; kc < 48; ++kc) {
      const int kb = kc * 64;
      v8i a[4];
#pragma unroll
      for (int mt = 0; mt < 4; ++mt) {
#pragma unroll
        for (int c = 0; c < 4; ++c) {
          int2 t = *(const int2*)(gS + (mt * 16 + m) * Fm + kb + c * 16 + hi * 8);
          a[mt][2 * c] = t.x;
          a[mt][2 * c + 1] = t.y;
        }
      }
#pragma unroll
      for (int i = 0; i < 3; ++i) {
        const int n0 = (tileBase + i) * 16;
        v8i bm;
#pragma unroll
        for (int c = 0; c < 4; ++c) {
          int2 t = *(const int2*)(w2t + (n0 + m) * Fm + kb + c * 16 + hi * 8);
          bm[2 * c] = t.x;
          bm[2 * c + 1] = t.y;
        }
#pragma unroll
        for (int mt = 0; mt < 4; ++mt)
          acc[i][mt] = __builtin_amdgcn_wmma_i32_16x16x64_iu8(
              true, a[mt], true, bm, acc[i][mt], false, false);
      }
    }
#pragma unroll
    for (int i = 0; i < 3; ++i) {
      const int col = (tileBase + i) * 16 + m;
      const float bias = b2[col] - (float)(ZP_G_OUT * colsum2[col]) * OUT_SCALE;
#pragma unroll
      for (int mt = 0; mt < 4; ++mt) {
#pragma unroll
        for (int j = 0; j < 8; ++j) {
          out[(rowBase + mt * 16 + j + 8 * hi) * Dm + col] =
              (float)acc[i][mt][j] * OUT_SCALE + bias;
        }
      }
    }
  }
}

extern "C" void kernel_launch(void* const* d_in, const int* in_sizes, int n_in,
                              void* d_out, int out_size, void* d_ws, size_t ws_size,
                              hipStream_t stream) {
  (void)in_sizes; (void)n_in; (void)out_size; (void)ws_size;
  const float* hidden = (const float*)d_in[0];
  const float* b2     = (const float*)d_in[1];
  const int*   W1     = (const int*)d_in[2];
  const int*   b1     = (const int*)d_in[3];
  const int*   W2     = (const int*)d_in[4];
  const int*   lut    = (const int*)d_in[5];
  float* out = (float*)d_out;

  signed char* w1t = (signed char*)d_ws;
  signed char* w2t = w1t + (size_t)Fm * Dm;
  int* colsum = (int*)(w2t + (size_t)Fm * Dm);

  const int nW = Fm * Dm;
  pack_w1t<<<nW / 256, 256, 0, stream>>>(W1, w1t);
  pack_w2t<<<nW / 256, 256, 0, stream>>>(W2, w2t);
  colsum_w2<<<Dm, 256, 0, stream>>>(W2, colsum);

  mlp_fused<<<NBLK, 256, SMEM_BYTES, stream>>>(hidden, b1, b2, lut, w1t, w2t,
                                               colsum, out);
}
